// GAT_25022479466661
// MI455X (gfx1250) — matrix operand order repaired and checked by reading, not compiled, not run on hardware
//
#include <hip/hip_runtime.h>
#include <hip/hip_bf16.h>
#include <math.h>

// ---------------------------------------------------------------------------
// CDNA5 (gfx1250) implementation. wave32. Matrix ops via
// v_wmma_f32_16x16x32_bf16 (D f32 16x16 = A bf16 16x32 * B bf16 32x16 + C).
//
// WMMA register layouts used (wave32; contiguous k, same order in A and B):
//   A (16x32 bf16): lane l -> row m=l&15, half h=l>>4;
//        elem i(0..15) = K = h*16 + i
//   B (32x16 bf16, column-major): lane l -> col n=l&15, half h=l>>4;
//        elem i(0..15) = K = h*16 + i
//   C/D (16x16 f32): lane l -> col n=l&15; VGPR v -> row m = (l>>4)*8 + v
// ---------------------------------------------------------------------------

typedef __attribute__((ext_vector_type(16))) __bf16 v16bf;
typedef __attribute__((ext_vector_type(8)))  __bf16 v8bf;
typedef __attribute__((ext_vector_type(4)))  __bf16 v4bf;
typedef __attribute__((ext_vector_type(8)))  float  v8f;

// problem constants (fixed by the reference)
constexpr int Bn = 8, Tn = 5, KD = 20, FTn = 384, FPn = 3, NS = 128;
constexpr int Hh = 64, HEADSn = 8, CLn = 2, Gg = 192;   // Gg = 3*H
constexpr int M1 = Bn * NS * Tn;   // 5120 sequences for GRU1
constexpr int M2 = Bn * NS;        // 1024 sequences for GRU2/3
constexpr float ALPHAv = 0.2f;
constexpr float NEGVv  = -9e15f;

__device__ __forceinline__ float sigm(float x) { return 1.f / (1.f + expf(-x)); }

__device__ __forceinline__ v8f wmma_bf16(v16bf a, v16bf b, v8f c) {
  // 8 args: (neg_a, A, neg_b, B, c_mod, C, reuse_a, reuse_b)
  return __builtin_amdgcn_wmma_f32_16x16x32_bf16(false, a, false, b,
                                                 (short)0, c, false, false);
}

// ---------------------------------------------------------------------------
// Fused GRU scan. One workgroup (4 waves / 128 thr) owns 64 sequences, keeps
// h in LDS, and runs the whole length-L recurrence. Per step:
//   gx = x_t @ Wi (+bi)  and  gh = h @ Wh (+bh)  via WMMA bf16, then gates.
// XM: 0 = x contiguous [M,L,F]; 1 = gather from s[B,T,KD,FT,NS] (block maps to
// (b, t, n0) so global loads are contiguous along n); 2 = gather from
// p[B,T,FP,NS].
// ---------------------------------------------------------------------------
template <int XM, int F, int L>
__global__ void __launch_bounds__(128)
k_gru(const float* __restrict__ xsrc,
      const __bf16* __restrict__ Wi, const __bf16* __restrict__ Wh,
      const float* __restrict__ bi, const float* __restrict__ bh,
      float* __restrict__ full)
{
  constexpr bool FULLK = (F % 32) == 0;   // no K-guards needed

  __shared__ __align__(32) __bf16 At[64][32];    // A tile (x) bf16
  __shared__ __align__(32) __bf16 Bt[Gg][32];    // B tile, transposed [col][k]
  __shared__ float                hS[64][Hh];    // h state f32
  __shared__ __align__(32) __bf16 hB[64][Hh];    // h state bf16 (A for gh)

  const int tid  = threadIdx.x;
  const int wave = tid >> 5;
  const int lane = tid & 31;
  const int lh   = lane >> 4;   // lane half
  const int lm   = lane & 15;

  int bb = 0, tt = 0, n0 = 0, m0 = 0;
  if (XM == 1) {
    const int g = blockIdx.x;            // Bn*Tn*2 = 80 blocks
    bb = g / (Tn * 2);
    const int rem = g % (Tn * 2);
    tt = rem >> 1;
    n0 = (rem & 1) * 64;
  } else {
    m0 = blockIdx.x * 64;
  }

  for (int e = tid; e < 64 * Hh; e += 128) {
    ((float*)hS)[e] = 0.f;
    ((__bf16*)hB)[e] = (__bf16)0.f;
  }
  __syncthreads();

  for (int step = 0; step < L; ++step) {
    v8f ax[12], ah[12];
#pragma unroll
    for (int j = 0; j < 12; ++j) {
      ax[j] = (v8f){0.f,0.f,0.f,0.f,0.f,0.f,0.f,0.f};
      ah[j] = (v8f){0.f,0.f,0.f,0.f,0.f,0.f,0.f,0.f};
    }

    // ---- gx = x_step @ Wi : K-loop over F in chunks of 32 ----
    for (int k0 = 0; k0 < F; k0 += 32) {
      // ---- A tile: vectorized float4 loads, no guards when F%32==0 ----
      if (XM == 0) {
        // contiguous rows: coalesce + vectorize along f
#pragma unroll
        for (int i = 0; i < 4; ++i) {            // 64*32 / (128*4)
          const int idx = tid + 128 * i;         // [0,512)
          const int r = idx >> 3, c4 = (idx & 7) * 4;
          const float4 v = *(const float4*)
              &xsrc[((size_t)(m0 + r) * L + step) * F + k0 + c4];
          v4bf o;
          o[0] = (__bf16)v.x; o[1] = (__bf16)v.y;
          o[2] = (__bf16)v.z; o[3] = (__bf16)v.w;
          *(v4bf*)&At[r][c4] = o;
        }
      } else if (XM == 1) {
        // gather from s: coalesce + vectorize along n (4 rows per load)
#pragma unroll
        for (int i = 0; i < 4; ++i) {
          const int idx = tid + 128 * i;         // [0,512)
          const int c = idx >> 4;                // 32 cols
          const int r = (idx & 15) * 4;          // rows 0,4,...,60
          const float4 v = *(const float4*)
              &xsrc[(((size_t)(bb * Tn + tt) * KD + step) * FTn + k0 + c) * NS
                    + n0 + r];
          At[r + 0][c] = (__bf16)v.x;
          At[r + 1][c] = (__bf16)v.y;
          At[r + 2][c] = (__bf16)v.z;
          At[r + 3][c] = (__bf16)v.w;
        }
      } else {
        // F=3 price path: tiny, keep guarded scalar loads
        for (int e = tid; e < 64 * 32; e += 128) {
          const int r = e & 63, c = e >> 6;
          const int f = k0 + c;
          float v = 0.f;
          if (f < F) {
            const int m = m0 + r, b2 = m >> 7, n2 = m & 127;
            v = xsrc[(((size_t)(b2 * Tn + step) * FPn + f) * NS + n2)];
          }
          At[r][c] = (__bf16)v;
        }
      }
      // ---- B tile: 16-byte bf16 loads, transpose on LDS store ----
#pragma unroll
      for (int i = 0; i < 6; ++i) {              // Gg*32/8 = 768 chunks
        const int idx = tid + 128 * i;           // [0,768)
        const int c8 = (idx % 24) * 8;
        const int kk = idx / 24;
        v8bf vb;
        if (FULLK || (k0 + kk) < F) {
          vb = *(const v8bf*)&Wi[(size_t)(k0 + kk) * Gg + c8];
        } else {
          vb = (v8bf){(__bf16)0.f,(__bf16)0.f,(__bf16)0.f,(__bf16)0.f,
                      (__bf16)0.f,(__bf16)0.f,(__bf16)0.f,(__bf16)0.f};
        }
#pragma unroll
        for (int q = 0; q < 8; ++q) Bt[c8 + q][kk] = vb[q];
      }
      __syncthreads();
      const int ar = wave * 16 + lm;
      const v8bf alo = *(const v8bf*)&At[ar][lh * 16];
      const v8bf ahi = *(const v8bf*)&At[ar][lh * 16 + 8];
      const v16bf a = __builtin_shufflevector(alo, ahi,
          0,1,2,3,4,5,6,7,8,9,10,11,12,13,14,15);
#pragma unroll
      for (int j = 0; j < 12; ++j) {
        const v16bf bf_ = *(const v16bf*)&Bt[j * 16 + lm][lh * 16];
        ax[j] = wmma_bf16(a, bf_, ax[j]);
      }
      __syncthreads();
    }

    // ---- gh = h @ Wh : K-loop over H=64 in chunks of 32 ----
    for (int k0 = 0; k0 < Hh; k0 += 32) {
#pragma unroll
      for (int i = 0; i < 6; ++i) {
        const int idx = tid + 128 * i;
        const int c8 = (idx % 24) * 8;
        const int kk = idx / 24;
        const v8bf vb = *(const v8bf*)&Wh[(size_t)(k0 + kk) * Gg + c8];
#pragma unroll
        for (int q = 0; q < 8; ++q) Bt[c8 + q][kk] = vb[q];
      }
      __syncthreads();
      const int ar = wave * 16 + lm;   // each wave touches only its 16 rows
      const v8bf alo = *(const v8bf*)&hB[ar][k0 + lh * 16];
      const v8bf ahi = *(const v8bf*)&hB[ar][k0 + lh * 16 + 8];
      const v16bf a = __builtin_shufflevector(alo, ahi,
          0,1,2,3,4,5,6,7,8,9,10,11,12,13,14,15);
#pragma unroll
      for (int j = 0; j < 12; ++j) {
        const v16bf bf_ = *(const v16bf*)&Bt[j * 16 + lm][lh * 16];
        ah[j] = wmma_bf16(a, bf_, ah[j]);
      }
      __syncthreads();
    }

    // ---- GRU gates (lane owns rows wave*16+lh*8+v, col j*16+lm) ----
#pragma unroll
    for (int j = 0; j < 4; ++j) {
      const int hc = j * 16 + lm;
      const float bir = bi[hc],          bhr = bh[hc];
      const float biz = bi[Hh + hc],     bhz = bh[Hh + hc];
      const float bin = bi[2 * Hh + hc], bhn = bh[2 * Hh + hc];
#pragma unroll
      for (int v = 0; v < 8; ++v) {
        const int rr = wave * 16 + lh * 8 + v;
        const float r = sigm((ax[j][v]     + bir) + (ah[j][v]     + bhr));
        const float z = sigm((ax[j + 4][v] + biz) + (ah[j + 4][v] + bhz));
        const float n = tanhf((ax[j + 8][v] + bin) + r * (ah[j + 8][v] + bhn));
        const float hnew = (1.f - z) * n + z * hS[rr][hc];
        hS[rr][hc] = hnew;
        hB[rr][hc] = (__bf16)hnew;
        size_t m;
        if (XM == 1) m = (size_t)(bb * NS + n0 + rr) * Tn + tt;
        else         m = (size_t)(m0 + rr);
        full[(m * L + step) * Hh + hc] = hnew;
      }
    }
    __syncthreads();
  }
}

// ---------------------------------------------------------------------------
// Temporal attention: a = softmax_l(<full[m,l,:], full[m,L-1,:]>),
// ctx[m,:] = sum_l a_l full[m,l,:].  One 64-thread block per row.
// ---------------------------------------------------------------------------
__global__ void __launch_bounds__(64)
k_attn(const float* __restrict__ full, float* __restrict__ ctx, int L)
{
  const int m = blockIdx.x, j = threadIdx.x;
  __shared__ float red[64];
  __shared__ float w[32];
  const float lastj = full[((size_t)m * L + (L - 1)) * Hh + j];
  for (int l = 0; l < L; ++l) {
    red[j] = full[((size_t)m * L + l) * Hh + j] * lastj;
    __syncthreads();
    if (j == 0) { float s = 0.f; for (int q = 0; q < 64; ++q) s += red[q]; w[l] = s; }
    __syncthreads();
  }
  if (j == 0) {
    float mx = w[0];
    for (int l = 1; l < L; ++l) mx = fmaxf(mx, w[l]);
    float s = 0.f;
    for (int l = 0; l < L; ++l) { w[l] = expf(w[l] - mx); s += w[l]; }
    const float inv = 1.f / s;
    for (int l = 0; l < L; ++l) w[l] *= inv;
  }
  __syncthreads();
  float acc = 0.f;
  for (int l = 0; l < L; ++l) acc += w[l] * full[((size_t)m * L + l) * Hh + j];
  ctx[(size_t)m * Hh + j] = acc;
}

// ---------------------------------------------------------------------------
// Generic WMMA GEMM: C[M,N] = A_f32[M,KT] * B_bf16[KT,N].  128 thr / 4 waves,
// 64x64 output tile per block, grid = (M/64, N/64). KT compile-time, %32==0.
// ---------------------------------------------------------------------------
template <int KT>
__global__ void __launch_bounds__(128)
k_gemm(const float* __restrict__ A, const __bf16* __restrict__ Bm,
       float* __restrict__ Cm, int Ndim)
{
  __shared__ __align__(32) __bf16 At[64][32];
  __shared__ __align__(32) __bf16 Bt[64][32];   // transposed [col][k]
  const int m0 = blockIdx.x * 64, n0 = blockIdx.y * 64;
  const int tid = threadIdx.x, wave = tid >> 5, lane = tid & 31;
  const int lh = lane >> 4, lm = lane & 15;
  v8f acc[4];
#pragma unroll
  for (int j = 0; j < 4; ++j) acc[j] = (v8f){0.f,0.f,0.f,0.f,0.f,0.f,0.f,0.f};

  for (int k0 = 0; k0 < KT; k0 += 32) {
#pragma unroll
    for (int i = 0; i < 4; ++i) {                // 64*32/(128*4)
      const int idx = tid + 128 * i;             // [0,512)
      const int r = idx >> 3, c4 = (idx & 7) * 4;
      const float4 v = *(const float4*)&A[(size_t)(m0 + r) * KT + k0 + c4];
      v4bf o;
      o[0] = (__bf16)v.x; o[1] = (__bf16)v.y;
      o[2] = (__bf16)v.z; o[3] = (__bf16)v.w;
      *(v4bf*)&At[r][c4] = o;
    }
#pragma unroll
    for (int i = 0; i < 2; ++i) {                // 32*64/8 = 256 chunks
      const int idx = tid + 128 * i;             // [0,256)
      const int kk = idx >> 3, c8 = (idx & 7) * 8;
      const v8bf vb = *(const v8bf*)&Bm[(size_t)(k0 + kk) * Ndim + n0 + c8];
#pragma unroll
      for (int q = 0; q < 8; ++q) Bt[c8 + q][kk] = vb[q];
    }
    __syncthreads();
    const int ar = wave * 16 + lm;
    const v8bf alo = *(const v8bf*)&At[ar][lh * 16];
    const v8bf ahi = *(const v8bf*)&At[ar][lh * 16 + 8];
    const v16bf a = __builtin_shufflevector(alo, ahi,
        0,1,2,3,4,5,6,7,8,9,10,11,12,13,14,15);
#pragma unroll
    for (int j = 0; j < 4; ++j) {
      const v16bf bf_ = *(const v16bf*)&Bt[j * 16 + lm][lh * 16];
      acc[j] = wmma_bf16(a, bf_, acc[j]);
    }
    __syncthreads();
  }
#pragma unroll
  for (int j = 0; j < 4; ++j)
#pragma unroll
    for (int v = 0; v < 8; ++v)
      Cm[(size_t)(m0 + wave * 16 + lh * 8 + v) * Ndim + n0 + j * 16 + lm] = acc[j][v];
}

// ---------------------------------------------------------------------------
// weight conversion / permutation kernels
// ---------------------------------------------------------------------------
__global__ void k_cvt(const float* __restrict__ src, __bf16* __restrict__ dst, int n)
{
  for (int i = blockIdx.x * blockDim.x + threadIdx.x; i < n;
       i += gridDim.x * blockDim.x)
    dst[i] = (__bf16)src[i];
}

// Wb [o,i,j] -> WbB [i, o*64+j]
__global__ void k_cvt_Wb(const float* __restrict__ src, __bf16* __restrict__ dst)
{
  const int idx = blockIdx.x * blockDim.x + threadIdx.x;
  if (idx >= Hh * Hh * Hh) return;
  const int i = idx >> 12, o = (idx >> 6) & 63, j = idx & 63;
  dst[idx] = (__bf16)src[((size_t)o * Hh + i) * Hh + j];
}

// W1 [k,h,f] -> W1c [h, k*64+f]
__global__ void k_cvt_W1(const float* __restrict__ src, __bf16* __restrict__ dst)
{
  const int idx = blockIdx.x * blockDim.x + threadIdx.x;
  if (idx >= Hh * HEADSn * Hh) return;
  const int h = idx >> 9, k = (idx >> 6) & 7, f = idx & 63;
  dst[idx] = (__bf16)src[((size_t)k * Hh + h) * Hh + f];
}

// ---------------------------------------------------------------------------
// bilinear finish: X[bn,o] = tanh(bb[o] + sum_j tmp[bn,o*64+j]*qt[bn,j])
// ---------------------------------------------------------------------------
__global__ void k_bil_finish(const float* __restrict__ tmp,
                             const float* __restrict__ qt,
                             const float* __restrict__ bbv,
                             float* __restrict__ X)
{
  const int idx = blockIdx.x * blockDim.x + threadIdx.x;
  if (idx >= M2 * Hh) return;
  const int bn = idx >> 6, o = idx & 63;
  const float* trow = tmp + (size_t)bn * (Hh * Hh) + o * Hh;
  const float* qrow = qt + (size_t)bn * Hh;
  float acc = bbv[o];
  for (int j = 0; j < Hh; ++j) acc += trow[j] * qrow[j];
  X[idx] = tanhf(acc);
}

// out1[bn,c] = tanh(bx[n,c] + sum_h X[bn,h]*Wx[n,c,h])
__global__ void k_out1(const float* __restrict__ X, const float* __restrict__ Wx,
                       const float* __restrict__ bx, float* __restrict__ out1)
{
  const int idx = blockIdx.x * blockDim.x + threadIdx.x;
  if (idx >= M2 * CLn) return;
  const int bn = idx >> 1, c = idx & 1, n = bn & (NS - 1);
  float acc = bx[n * CLn + c];
  const float* w = Wx + ((size_t)n * CLn + c) * Hh;
  const float* x = X + (size_t)bn * Hh;
  for (int h = 0; h < Hh; ++h) acc += x[h] * w[h];
  out1[idx] = tanhf(acc);
}

// el/er logits for GAT layer 1: Whd stored [b,n, k*64+f]
__global__ void k_gat1_logits(const float* __restrict__ Whd,
                              const float* __restrict__ a1,
                              float* __restrict__ el, float* __restrict__ er)
{
  const int idx = blockIdx.x * blockDim.x + threadIdx.x;
  if (idx >= Bn * HEADSn * NS) return;
  const int b = idx >> 10, k = (idx >> 7) & 7, n = idx & 127;
  const float* row = Whd + ((size_t)(b * NS + n)) * (HEADSn * Hh) + k * Hh;
  float e1 = 0.f, e2 = 0.f;
  for (int f = 0; f < Hh; ++f) {
    e1 += row[f] * a1[k * (2 * Hh) + f];
    e2 += row[f] * a1[k * (2 * Hh) + Hh + f];
  }
  el[idx] = e1;
  er[idx] = e2;
}

// masked softmax over neighbors + aggregation + ELU -> X2[b,n,k*64+f]
__global__ void __launch_bounds__(128)
k_gat1_apply(const float* __restrict__ el, const float* __restrict__ er,
             const float* __restrict__ adj, const float* __restrict__ Whd,
             float* __restrict__ X2)
{
  const int b = blockIdx.x >> 10, k = (blockIdx.x >> 7) & 7, n = blockIdx.x & 127;
  const int m = threadIdx.x;
  __shared__ float att[128];
  float e = el[(b * HEADSn + k) * NS + n] + er[(b * HEADSn + k) * NS + m];
  e = (e > 0.f) ? e : ALPHAv * e;
  att[m] = (adj[n * NS + m] > 0.f) ? e : NEGVv;
  __syncthreads();
  if (m == 0) {
    float mx = att[0];
    for (int i = 1; i < NS; ++i) mx = fmaxf(mx, att[i]);
    float s = 0.f;
    for (int i = 0; i < NS; ++i) { const float t = expf(att[i] - mx); att[i] = t; s += t; }
    const float inv = 1.f / s;
    for (int i = 0; i < NS; ++i) att[i] *= inv;
  }
  __syncthreads();
  if (m < Hh) {
    float acc = 0.f;
    for (int mm = 0; mm < NS; ++mm)
      acc += att[mm] * Whd[(size_t)(b * NS + mm) * (HEADSn * Hh) + k * Hh + m];
    acc = (acc > 0.f) ? acc : expf(acc) - 1.f;   // ELU
    X2[(size_t)(b * NS + n) * (HEADSn * Hh) + k * Hh + m] = acc;
  }
}

// Wh2[bn,c] = sum_q X2[bn,q]*W2[q,c]
__global__ void k_gat2_wh2(const float* __restrict__ X2, const float* __restrict__ W2,
                           float* __restrict__ Wh2m)
{
  const int idx = blockIdx.x * blockDim.x + threadIdx.x;
  if (idx >= M2 * CLn) return;
  const int bn = idx >> 1, c = idx & 1;
  const float* x = X2 + (size_t)bn * (HEADSn * Hh);
  float acc = 0.f;
  for (int q = 0; q < HEADSn * Hh; ++q) acc += x[q] * W2[q * CLn + c];
  Wh2m[idx] = acc;
}

// layer-2 masked softmax + aggregation, add out1, transpose-store to d_out
__global__ void __launch_bounds__(128)
k_gat2_final(const float* __restrict__ Wh2m, const float* __restrict__ a2,
             const float* __restrict__ adj, const float* __restrict__ out1,
             float* __restrict__ dout)
{
  const int b = blockIdx.x >> 7, n = blockIdx.x & 127;
  const int m = threadIdx.x;
  __shared__ float att[128];
  const float eln = Wh2m[(b * NS + n) * CLn] * a2[0] +
                    Wh2m[(b * NS + n) * CLn + 1] * a2[1];
  const float erm = Wh2m[(b * NS + m) * CLn] * a2[2] +
                    Wh2m[(b * NS + m) * CLn + 1] * a2[3];
  float e = eln + erm;
  e = (e > 0.f) ? e : ALPHAv * e;
  att[m] = (adj[n * NS + m] > 0.f) ? e : NEGVv;
  __syncthreads();
  if (m == 0) {
    float mx = att[0];
    for (int i = 1; i < NS; ++i) mx = fmaxf(mx, att[i]);
    float s = 0.f;
    for (int i = 0; i < NS; ++i) { const float t = expf(att[i] - mx); att[i] = t; s += t; }
    const float inv = 1.f / s;
    for (int i = 0; i < NS; ++i) att[i] *= inv;
  }
  __syncthreads();
  if (m < CLn) {
    float acc = 0.f;
    for (int mm = 0; mm < NS; ++mm) acc += att[mm] * Wh2m[(b * NS + mm) * CLn + m];
    dout[((size_t)b * CLn + m) * NS + n] = acc + out1[(b * NS + n) * CLn + m];
  }
}

// ---------------------------------------------------------------------------
extern "C" void kernel_launch(void* const* d_in, const int* in_sizes, int n_in,
                              void* d_out, int out_size, void* d_ws, size_t ws_size,
                              hipStream_t stream)
{
  (void)in_sizes; (void)n_in; (void)out_size; (void)ws_size;
  const float* s      = (const float*)d_in[0];
  const float* p      = (const float*)d_in[1];
  const float* adj    = (const float*)d_in[2];
  const float* sm1_Wi = (const float*)d_in[3];
  const float* sm1_Wh = (const float*)d_in[4];
  const float* sm1_bi = (const float*)d_in[5];
  const float* sm1_bh = (const float*)d_in[6];
  const float* sm2_Wi = (const float*)d_in[7];
  const float* sm2_Wh = (const float*)d_in[8];
  const float* sm2_bi = (const float*)d_in[9];
  const float* sm2_bh = (const float*)d_in[10];
  const float* pr_Wi  = (const float*)d_in[11];
  const float* pr_Wh  = (const float*)d_in[12];
  const float* pr_bi  = (const float*)d_in[13];
  const float* pr_bh  = (const float*)d_in[14];
  const float* Wb     = (const float*)d_in[15];
  const float* bbv    = (const float*)d_in[16];
  const float* Wx     = (const float*)d_in[17];
  const float* bx     = (const float*)d_in[18];
  const float* W1     = (const float*)d_in[19];
  const float* a1     = (const float*)d_in[20];
  const float* W2     = (const float*)d_in[21];
  const float* a2     = (const float*)d_in[22];

  char* wsb = (char*)d_ws;
  size_t off = 0;
  auto alloc = [&](size_t bytes) -> char* {
    char* r = wsb + off;
    off = (off + bytes + 255) & ~(size_t)255;
    return r;
  };

  __bf16* bWi1 = (__bf16*)alloc((size_t)FTn * Gg * 2);
  __bf16* bWh1 = (__bf16*)alloc((size_t)Hh * Gg * 2);
  __bf16* bWi2 = (__bf16*)alloc((size_t)Hh * Gg * 2);
  __bf16* bWh2 = (__bf16*)alloc((size_t)Hh * Gg * 2);
  __bf16* bWi3 = (__bf16*)alloc((size_t)FPn * Gg * 2);
  __bf16* bWh3 = (__bf16*)alloc((size_t)Hh * Gg * 2);
  __bf16* bWbB = (__bf16*)alloc((size_t)Hh * Hh * Hh * 2);
  __bf16* bW1c = (__bf16*)alloc((size_t)Hh * HEADSn * Hh * 2);
  float* full1 = (float*)alloc((size_t)M1 * KD * Hh * 4);
  float* cv    = (float*)alloc((size_t)M1 * Hh * 4);
  float* full2 = (float*)alloc((size_t)M2 * Tn * Hh * 4);
  float* ct    = (float*)alloc((size_t)M2 * Hh * 4);
  float* full3 = (float*)alloc((size_t)M2 * Tn * Hh * 4);
  float* qt    = (float*)alloc((size_t)M2 * Hh * 4);
  float* tmpB  = (float*)alloc((size_t)M2 * Hh * Hh * 4);
  float* Xb    = (float*)alloc((size_t)M2 * Hh * 4);
  float* out1  = (float*)alloc((size_t)M2 * CLn * 4);
  float* Whd   = (float*)alloc((size_t)M2 * HEADSn * Hh * 4);
  float* el    = (float*)alloc((size_t)Bn * HEADSn * NS * 4);
  float* er    = (float*)alloc((size_t)Bn * HEADSn * NS * 4);
  float* X2    = (float*)alloc((size_t)M2 * HEADSn * Hh * 4);
  float* Wh2m  = (float*)alloc((size_t)M2 * CLn * 4);

  auto cv_launch = [&](const float* src, __bf16* dst, int n) {
    k_cvt<<<(n + 255) / 256, 256, 0, stream>>>(src, dst, n);
  };
  cv_launch(sm1_Wi, bWi1, FTn * Gg);
  cv_launch(sm1_Wh, bWh1, Hh * Gg);
  cv_launch(sm2_Wi, bWi2, Hh * Gg);
  cv_launch(sm2_Wh, bWh2, Hh * Gg);
  cv_launch(pr_Wi,  bWi3, FPn * Gg);
  cv_launch(pr_Wh,  bWh3, Hh * Gg);
  k_cvt_Wb<<<(Hh * Hh * Hh + 255) / 256, 256, 0, stream>>>(Wb, bWbB);
  k_cvt_W1<<<(Hh * HEADSn * Hh + 255) / 256, 256, 0, stream>>>(W1, bW1c);

  // SMEncoder stage 1: GRU over 20 days of FT=384 features (dominant GEMM)
  k_gru<1, FTn, KD><<<Bn * Tn * 2, 128, 0, stream>>>(s, bWi1, bWh1,
                                                     sm1_bi, sm1_bh, full1);
  k_attn<<<M1, 64, 0, stream>>>(full1, cv, KD);
  // SMEncoder stage 2: GRU over T=5 contexts
  k_gru<0, Hh, Tn><<<M2 / 64, 128, 0, stream>>>(cv, bWi2, bWh2,
                                                sm2_bi, sm2_bh, full2);
  k_attn<<<M2, 64, 0, stream>>>(full2, ct, Tn);
  // PriceEncoder
  k_gru<2, FPn, Tn><<<M2 / 64, 128, 0, stream>>>(p, bWi3, bWh3,
                                                 pr_bi, pr_bh, full3);
  k_attn<<<M2, 64, 0, stream>>>(full3, qt, Tn);

  // bilinear fusion: tmp = ct @ WbB  ->  X = tanh(tmp . qt + bb)
  k_gemm<Hh><<<dim3(M2 / 64, (Hh * Hh) / 64), 128, 0, stream>>>(ct, bWbB, tmpB,
                                                                Hh * Hh);
  k_bil_finish<<<(M2 * Hh + 255) / 256, 256, 0, stream>>>(tmpB, qt, bbv, Xb);
  k_out1<<<(M2 * CLn + 255) / 256, 256, 0, stream>>>(Xb, Wx, bx, out1);

  // GAT layer 1
  k_gemm<Hh><<<dim3(M2 / 64, (HEADSn * Hh) / 64), 128, 0, stream>>>(
      Xb, bW1c, Whd, HEADSn * Hh);
  k_gat1_logits<<<(Bn * HEADSn * NS + 255) / 256, 256, 0, stream>>>(Whd, a1, el, er);
  k_gat1_apply<<<Bn * HEADSn * NS, 128, 0, stream>>>(el, er, adj, Whd, X2);

  // GAT layer 2 + residual + final transpose store
  k_gat2_wh2<<<(M2 * CLn + 255) / 256, 256, 0, stream>>>(X2, W2, Wh2m);
  k_gat2_final<<<M2, 128, 0, stream>>>(Wh2m, a2, adj, out1, (float*)d_out);
}
